// MultiHeadAttention_11046655885676
// MI455X (gfx1250) — matrix operand order repaired and checked by reading, not compiled, not run on hardware
//
#include <hip/hip_runtime.h>

// ---------------- common types / helpers ----------------
typedef __attribute__((ext_vector_type(16))) __bf16 v16bf;
typedef __attribute__((ext_vector_type(8)))  __bf16 v8bf;
typedef __attribute__((ext_vector_type(8)))  float  v8f;

// stage 8 contiguous elements into LDS as bf16 (fp32 source: convert)
__device__ __forceinline__ void stage8(const float* __restrict__ g, __bf16* s) {
    const float4 a = *(const float4*)g;
    const float4 b = *(const float4*)(g + 4);
    v8bf r;
    r[0] = (__bf16)a.x; r[1] = (__bf16)a.y; r[2] = (__bf16)a.z; r[3] = (__bf16)a.w;
    r[4] = (__bf16)b.x; r[5] = (__bf16)b.y; r[6] = (__bf16)b.z; r[7] = (__bf16)b.w;
    *(v8bf*)s = r;
}
__device__ __forceinline__ void stage8(const __bf16* __restrict__ g, __bf16* s) {
    *(v8bf*)s = *(const v8bf*)g;
}

// build a v16bf fragment from two aligned 16-byte LDS reads
__device__ __forceinline__ v16bf ld_frag2(const __bf16* p0, const __bf16* p1) {
    v8bf lo = *(const v8bf*)p0;
    v8bf hi = *(const v8bf*)p1;
    v16bf r;
#pragma unroll
    for (int i = 0; i < 8; ++i) { r[i] = lo[i]; r[i + 8] = hi[i]; }
    return r;
}

// problem constants
static constexpr int Bz  = 4;
static constexpr int Ntok = 2048;
static constexpr int C   = 1024;
static constexpr int H   = 16;
static constexpr int DHc = 64;
static constexpr float SCALE = 0.125f;   // 64^-0.5

// ---------------- Tensor Data Mover (gfx1250 TDM) ----------------
#if defined(__gfx1250__) && __has_builtin(__builtin_amdgcn_tensor_load_to_lds) && \
    __has_builtin(__builtin_amdgcn_s_wait_tensorcnt)
#define ATTN_USE_TDM 1
#else
#define ATTN_USE_TDM 0
#endif

#if ATTN_USE_TDM
typedef __attribute__((ext_vector_type(4))) unsigned int v4u32;
typedef __attribute__((ext_vector_type(8))) int          v8i32;
typedef __attribute__((ext_vector_type(4))) int          v4i32;

// DMA a [tile_rows x 64] bf16 tile (row stride 64 elems in global) into LDS,
// with pad-on-load: +4 DWORDs after every 32 DWORDs (128B row) -> LDS row
// stride of 72 bf16 elems, matching the padded fragment layout.
__device__ __forceinline__ void tdm_load_tile64_bf16(
    unsigned lds_off, const __bf16* g, unsigned tile_rows, unsigned tensor_rows)
{
    const unsigned long long ga = (unsigned long long)(size_t)g;
    v4u32 g0;
    g0[0] = 1u;                                                 // count=1 (valid), user mode
    g0[1] = lds_off;                                            // lds_addr (bytes)
    g0[2] = (unsigned)ga;                                       // global_addr[31:0]
    g0[3] = (unsigned)((ga >> 32) & 0x01FFFFFFu) | (2u << 30);  // global_addr[56:32] | type=2
    v8i32 g1;
    // data_size=1 (2B), pad_enable=1, pad_interval code 4 (=32 DWORDs), pad_amount code 3 (=4 DWORDs)
    g1[0] = (1 << 16) | (1 << 20) | (4 << 22) | (3 << 25);
    g1[1] = (int)(64u << 16);                                   // tensor_dim0=64 (bits 79:48)
    g1[2] = (int)((tensor_rows & 0xFFFFu) << 16);               // tensor_dim1 lo (bits 95:80)
    g1[3] = (int)(((tensor_rows >> 16) & 0xFFFFu) |             // tensor_dim1 hi (bits 111:96)
                  (64u << 16));                                 // tile_dim0=64 (bits 127:112)
    g1[4] = (int)(tile_rows & 0xFFFFu);                         // tile_dim1 (bits 143:128); tile_dim2=0
    g1[5] = 64;                                                 // tensor_dim0_stride=64 (bits 207:160)
    g1[6] = 0;
    g1[7] = 0;                                                  // tensor_dim1_stride=0
    const v4i32 z4 = {0, 0, 0, 0};
    const v8i32 z8 = {0, 0, 0, 0, 0, 0, 0, 0};
    __builtin_amdgcn_tensor_load_to_lds(g0, g1, z4, z4, z8, 0);
}
#endif

// ---------------- GEMM: Y = A @ W^T (+bias), bf16 WMMA, fp32 accum ----------------
// MODE 0: write fp32 out[m*N+n] = acc + bias[n]
// MODE 1: QKV scatter: bf16 into q/k/v workspaces laid out [B,H,N,DH]
static constexpr int BM = 128, BN = 128, BK = 32;
static constexpr int SAB = BK + 8;   // padded LDS stride (bf16 elems); 80B rows, 16B aligned

template <int MODE, typename AT>
__global__ __launch_bounds__(256) void gemm_bf16_wmma(
    const AT* __restrict__ A, const float* __restrict__ W,
    const float* __restrict__ bias, float* __restrict__ outF,
    __bf16* __restrict__ qw, __bf16* __restrict__ kw, __bf16* __restrict__ vw,
    int M, int N, int K)
{
    __shared__ __align__(16) __bf16 As[BM * SAB];
    __shared__ __align__(16) __bf16 Bs[BN * SAB];

    const int tid  = threadIdx.x;
    const int lane = tid & 31;
    const int wave = tid >> 5;
    const int wr   = wave >> 1;          // 0..3  -> 32-row strip
    const int wc   = wave & 1;           // 0..1  -> 64-col strip
    const int half = lane >> 4;
    const int l16  = lane & 15;
    const int mblk = blockIdx.y * BM;
    const int nblk = blockIdx.x * BN;

    v8f acc[2][4];
#pragma unroll
    for (int tr = 0; tr < 2; ++tr)
#pragma unroll
        for (int tc = 0; tc < 4; ++tc)
#pragma unroll
            for (int i = 0; i < 8; ++i) acc[tr][tc][i] = 0.f;

    for (int k0 = 0; k0 < K; k0 += BK) {
        // ---- stage A tile (BM x BK) and B tile (BN x BK, from W rows) ----
#pragma unroll
        for (int t = 0; t < 2; ++t) {
            const int ch  = tid + t * 256;       // 512 chunks of 8
            const int row = ch >> 2;             // 4 chunks per row
            const int kc  = (ch & 3) << 3;
            stage8(&A[(size_t)(mblk + row) * K + k0 + kc], &As[row * SAB + kc]);
            stage8(&W[(size_t)(nblk + row) * K + k0 + kc], &Bs[row * SAB + kc]);
        }
        if (k0 + BK < K) {   // prefetch next K-slab -> global_prefetch_b8
            const int row = tid >> 2, kc = (tid & 3) << 3;
            __builtin_prefetch(&A[(size_t)(mblk + row) * K + k0 + BK + kc], 0, 0);
            __builtin_prefetch(&W[(size_t)(nblk + row) * K + k0 + BK + kc], 0, 0);
        }
        __syncthreads();

        // ---- fragments + 8 WMMAs per wave ----
        v16bf af[2], bfr[4];
#pragma unroll
        for (int tr = 0; tr < 2; ++tr) {
            const int r = wr * 32 + tr * 16 + l16;
            af[tr] = ld_frag2(&As[r * SAB + 8 * half],
                              &As[r * SAB + 16 + 8 * half]);
        }
#pragma unroll
        for (int tc = 0; tc < 4; ++tc) {
            const int c = wc * 64 + tc * 16 + l16;
            bfr[tc] = ld_frag2(&Bs[c * SAB + 8 * half],
                               &Bs[c * SAB + 16 + 8 * half]);
        }
#pragma unroll
        for (int tr = 0; tr < 2; ++tr)
#pragma unroll
            for (int tc = 0; tc < 4; ++tc)
                acc[tr][tc] = __builtin_amdgcn_wmma_f32_16x16x32_bf16(
                    false, af[tr], false, bfr[tc], (short)0, acc[tr][tc], false, false);
        __syncthreads();
    }

    // ---- epilogue ----
#pragma unroll
    for (int tc = 0; tc < 4; ++tc) {
        const int colg = nblk + wc * 64 + tc * 16 + l16;
        const float bv = bias[colg];
#pragma unroll
        for (int tr = 0; tr < 2; ++tr) {
#pragma unroll
            for (int i = 0; i < 8; ++i) {
                const int rowg = mblk + wr * 32 + tr * 16 + half * 8 + i;
                const float val = acc[tr][tc][i] + bv;
                if constexpr (MODE == 0) {
                    outF[(size_t)rowg * N + colg] = val;
                } else {
                    const int which = colg >> 10;          // /C
                    const int rem   = colg & (C - 1);
                    const int h     = rem >> 6;            // /DH
                    const int d     = rem & (DHc - 1);
                    const int bb    = rowg >> 11;          // /Ntok
                    const int tok   = rowg & (Ntok - 1);
                    const size_t dst =
                        (((size_t)bb * H + h) * Ntok + tok) * DHc + d;
                    const __bf16 bval = (__bf16)val;
                    if (which == 0)      qw[dst] = bval;
                    else if (which == 1) kw[dst] = bval;
                    else                 vw[dst] = bval;
                }
            }
        }
    }
}

// ---------------- Flash attention: per (b,h), 128 queries/block ----------------
static constexpr int QB = 128;   // queries per block (8 waves x 16 rows)
static constexpr int KB = 64;    // keys per inner iteration
static constexpr int SQ  = DHc + 8;   // 72
static constexpr int SK  = DHc + 8;   // 72
static constexpr int SVt = KB + 8;    // 72
static constexpr int SP  = KB + 8;    // 72

__global__ __launch_bounds__(256) void attn_fa(
    const __bf16* __restrict__ q, const __bf16* __restrict__ k,
    const __bf16* __restrict__ v, __bf16* __restrict__ ao)
{
    __shared__ __align__(16) __bf16 Qs[QB * SQ];
    __shared__ __align__(16) __bf16 Ks[KB * SK];
    __shared__ __align__(16) __bf16 Vt[DHc * SVt];   // V transposed: [d][key]
    __shared__ __align__(16) __bf16 Ps[QB * SP];     // per-wave 16 rows

    const int tid  = threadIdx.x;
    const int lane = tid & 31;
    const int wave = tid >> 5;
    const int half = lane >> 4;
    const int l16  = lane & 15;
    const int bh   = blockIdx.y;
    const int bb   = bh >> 4;
    const int h    = bh & 15;
    const int m0   = blockIdx.x * QB;
    const size_t base = (size_t)bh * Ntok * DHc;

    // ---- stage Q block once (128 x 64): TDM on wave 0 ----
#if ATTN_USE_TDM
    const unsigned qs_off = (unsigned)(size_t)&Qs[0];
    const unsigned ks_off = (unsigned)(size_t)&Ks[0];
    if (wave == 0)
        tdm_load_tile64_bf16(qs_off, &q[base + (size_t)m0 * DHc], QB, Ntok);
#else
#pragma unroll
    for (int t = 0; t < 4; ++t) {
        const int ch  = tid + t * 256;
        const int row = ch >> 3;
        const int dc  = (ch & 7) << 3;
        stage8(&q[base + (size_t)(m0 + row) * DHc + dc], &Qs[row * SQ + dc]);
    }
#endif

    v8f Oa[4];
    float mprev[8], lsum[8];
#pragma unroll
    for (int t = 0; t < 4; ++t)
#pragma unroll
        for (int i = 0; i < 8; ++i) Oa[t][i] = 0.f;
#pragma unroll
    for (int i = 0; i < 8; ++i) { mprev[i] = -1e30f; lsum[i] = 0.f; }

    for (int kb0 = 0; kb0 < Ntok; kb0 += KB) {
        // ---- stage K via TDM (wave 0); V transposed manually (all waves) ----
#if ATTN_USE_TDM
        if (wave == 0)
            tdm_load_tile64_bf16(ks_off, &k[base + (size_t)kb0 * DHc], KB, Ntok);
#pragma unroll
        for (int t = 0; t < 2; ++t) {
            const int ch  = tid + t * 256;
            const int row = ch >> 3;
            const int dc  = (ch & 7) << 3;
            v8bf tv = *(const v8bf*)&v[base + (size_t)(kb0 + row) * DHc + dc];
#pragma unroll
            for (int j = 0; j < 8; ++j) Vt[(dc + j) * SVt + row] = tv[j];
        }
        if (wave == 0) __builtin_amdgcn_s_wait_tensorcnt(0);
#else
#pragma unroll
        for (int t = 0; t < 2; ++t) {
            const int ch  = tid + t * 256;
            const int row = ch >> 3;
            const int dc  = (ch & 7) << 3;
            stage8(&k[base + (size_t)(kb0 + row) * DHc + dc], &Ks[row * SK + dc]);
            v8bf tv = *(const v8bf*)&v[base + (size_t)(kb0 + row) * DHc + dc];
#pragma unroll
            for (int j = 0; j < 8; ++j) Vt[(dc + j) * SVt + row] = tv[j];
        }
#endif
        __syncthreads();

        // ---- S = Q K^T for this wave's 16 rows (4 key tiles) ----
        v8f Sacc[4];
#pragma unroll
        for (int t = 0; t < 4; ++t)
#pragma unroll
            for (int i = 0; i < 8; ++i) Sacc[t][i] = 0.f;
#pragma unroll
        for (int ks = 0; ks < DHc; ks += 32) {
            const int r = wave * 16 + l16;
            v16bf aq = ld_frag2(&Qs[r * SQ + ks + 8 * half],
                                &Qs[r * SQ + ks + 16 + 8 * half]);
#pragma unroll
            for (int tc = 0; tc < 4; ++tc) {
                const int c = tc * 16 + l16;
                v16bf bk = ld_frag2(&Ks[c * SK + ks + 8 * half],
                                    &Ks[c * SK + ks + 16 + 8 * half]);
                Sacc[tc] = __builtin_amdgcn_wmma_f32_16x16x32_bf16(
                    false, aq, false, bk, (short)0, Sacc[tc], false, false);
            }
        }

        // ---- online softmax (rows fully owned by this wave) ----
#pragma unroll
        for (int i = 0; i < 8; ++i) {
            float mloc = Sacc[0][i];
            mloc = fmaxf(mloc, Sacc[1][i]);
            mloc = fmaxf(mloc, Sacc[2][i]);
            mloc = fmaxf(mloc, Sacc[3][i]);
#pragma unroll
            for (int msk = 1; msk < 16; msk <<= 1)
                mloc = fmaxf(mloc, __shfl_xor(mloc, msk, 32));
            mloc *= SCALE;
            const float mnew  = fmaxf(mprev[i], mloc);
            const float alpha = __expf(mprev[i] - mnew);
            mprev[i] = mnew;
            float rs = 0.f;
            const int prow = wave * 16 + half * 8 + i;
#pragma unroll
            for (int tc = 0; tc < 4; ++tc) {
                const float p = __expf(Sacc[tc][i] * SCALE - mnew);
                rs += p;
                Ps[prow * SP + tc * 16 + l16] = (__bf16)p;
            }
#pragma unroll
            for (int msk = 1; msk < 16; msk <<= 1)
                rs += __shfl_xor(rs, msk, 32);
            lsum[i] = lsum[i] * alpha + rs;
#pragma unroll
            for (int td = 0; td < 4; ++td) Oa[td][i] *= alpha;
        }

        // ---- O += P @ V ----
#pragma unroll
        for (int ks = 0; ks < KB; ks += 32) {
            const int r = wave * 16 + l16;
            v16bf ap = ld_frag2(&Ps[r * SP + ks + 8 * half],
                                &Ps[r * SP + ks + 16 + 8 * half]);
#pragma unroll
            for (int td = 0; td < 4; ++td) {
                const int c = td * 16 + l16;   // d index
                v16bf bv = ld_frag2(&Vt[c * SVt + ks + 8 * half],
                                    &Vt[c * SVt + ks + 16 + 8 * half]);
                Oa[td] = __builtin_amdgcn_wmma_f32_16x16x32_bf16(
                    false, ap, false, bv, (short)0, Oa[td], false, false);
            }
        }
        __syncthreads();
    }

    // ---- normalize and write bf16 [b, tok, h*DH + d] for out-proj GEMM ----
#pragma unroll
    for (int i = 0; i < 8; ++i) {
        const float inv = 1.0f / lsum[i];
        const int tok = m0 + wave * 16 + half * 8 + i;
        const size_t rowbase = ((size_t)bb * Ntok + tok) * C + h * DHc;
#pragma unroll
        for (int td = 0; td < 4; ++td)
            ao[rowbase + td * 16 + l16] = (__bf16)(Oa[td][i] * inv);
    }
}

// ---------------- launch ----------------
extern "C" void kernel_launch(void* const* d_in, const int* in_sizes, int n_in,
                              void* d_out, int out_size, void* d_ws, size_t ws_size,
                              hipStream_t stream) {
    const float* x     = (const float*)d_in[0];
    const float* w_qkv = (const float*)d_in[1];
    const float* b_qkv = (const float*)d_in[2];
    const float* w_out = (const float*)d_in[3];
    const float* b_out = (const float*)d_in[4];
    float* out = (float*)d_out;

    const size_t HEAD_ELEMS = (size_t)Bz * H * Ntok * DHc;   // 8388608
    __bf16* wsb = (__bf16*)d_ws;
    __bf16* qw = wsb;
    __bf16* kw = wsb + HEAD_ELEMS;
    __bf16* vw = wsb + 2 * HEAD_ELEMS;
    __bf16* ao = wsb + 3 * HEAD_ELEMS;

    const int M = Bz * Ntok;   // 8192
    dim3 blk(256);

    // 1) QKV projection: [8192,1024] x [1024,3072] -> bf16 q/k/v [B,H,N,DH]
    gemm_bf16_wmma<1, float><<<dim3((3 * C) / BN, M / BM), blk, 0, stream>>>(
        x, w_qkv, b_qkv, nullptr, qw, kw, vw, M, 3 * C, C);

    // 2) Flash attention per (b,h): -> bf16 [b, tok, C]
    attn_fa<<<dim3(Ntok / QB, Bz * H), blk, 0, stream>>>(qw, kw, vw, ao);

    // 3) Output projection: [8192,1024] x [1024,1024] -> fp32 d_out
    gemm_bf16_wmma<0, __bf16><<<dim3(C / BN, M / BM), blk, 0, stream>>>(
        ao, w_out, b_out, out, nullptr, nullptr, nullptr, M, C, C);
}
